// Encoder_86517821213983
// MI455X (gfx1250) — matrix operand order repaired and checked by reading, not compiled, not run on hardware
//
#include <hip/hip_runtime.h>
#include <hip/hip_bf16.h>

typedef __attribute__((ext_vector_type(16))) __bf16          v16bf;
typedef __attribute__((ext_vector_type(16))) unsigned short  v16us;
typedef __attribute__((ext_vector_type(8)))  float           v8f;

struct U32x8 { uint4 lo; uint4 hi; };   // 32B, matches v16us size

__device__ __forceinline__ unsigned short f32_to_bf16_rne(float f) {
    unsigned int u = __float_as_uint(f);
    unsigned int lsb = (u >> 16) & 1u;
    u += 0x7FFFu + lsb;                 // round-to-nearest-even
    return (unsigned short)(u >> 16);
}
__device__ __forceinline__ float bf16_to_f32(unsigned short h) {
    return __uint_as_float(((unsigned int)h) << 16);
}

// ---------------------------------------------------------------------------
// Weight pre-swizzle: f32 (27,Cin,Cout) -> bf16 in WMMA B-fragment order.
// [k][cc][nt][lane][e]: element e of lane = (K = (e>>3)*16 + (lane>>4)*8 + (e&7),
// col = nt*16 + (lane&15)).  Conv loads B as 2 coalesced b128 per fragment.
// ---------------------------------------------------------------------------
__global__ void prep_weights(const float* __restrict__ W, unsigned short* __restrict__ Wb,
                             int Cin, int Cout, int KSTEPS, int NT, long long total) {
    long long idx = (long long)blockIdx.x * blockDim.x + threadIdx.x;
    if (idx >= total) return;
    int e    = (int)(idx & 15);
    int lane = (int)((idx >> 4) & 31);
    long long t = idx >> 9;
    int nt = (int)(t % NT);  t /= NT;
    int cc = (int)(t % KSTEPS);
    int k  = (int)(t / KSTEPS);
    int ci = (cc << 5) + ((e >> 3) << 4) + ((lane >> 4) << 3) + (e & 7);
    int co = (nt << 4) + (lane & 15);
    float v = (ci < Cin) ? W[((size_t)k * Cin + ci) * Cout + co] : 0.f;
    Wb[idx] = f32_to_bf16_rne(v);
}

__global__ void f32_to_bf16_vec(const float* __restrict__ src, unsigned short* __restrict__ dst, int n) {
    int i = blockIdx.x * blockDim.x + threadIdx.x;
    if (i < n) dst[i] = f32_to_bf16_rne(src[i]);
}

// ---------------------------------------------------------------------------
// Sparse conv: Y[n,:] = sum_k Xb[nbr[k,n],:] @ W[k]  (bf16 WMMA, f32 acc)
// One wave = one 16-row x 32-col output tile (2 accumulators).  The WMMA
// A-fragment for lane (m, half) is two contiguous 16B chunks of gathered row
// nbr[k,row0+m]: K = half*8..+7 and 16+half*8..+7  -> direct global_load_b128,
// no LDS, no barriers.  Tiles flattened over (rowTiles x colGroups) so every
// Cout keeps all waves busy.
// ---------------------------------------------------------------------------
__global__ __launch_bounds__(256) void sparse_conv_wmma(
    const unsigned short* __restrict__ Xb,   // (N_in, Cin) bf16
    const unsigned short* __restrict__ Wb,   // swizzled weights
    const int*            __restrict__ nbr,  // (27, N_out)
    float*                __restrict__ Y,    // (N_out, Cout)
    int N_out, int Cin, int Cout, int rowTiles, int colGroups)
{
    const int lane = threadIdx.x & 31;
    const int wave = threadIdx.x >> 5;
    const int wg   = blockIdx.x * 8 + wave;
    if (wg >= rowTiles * colGroups) return;

    const int rowTile  = wg / colGroups;
    const int colGroup = wg % colGroups;
    const int row0 = rowTile << 4;
    const int nt0  = colGroup << 1;
    const int col0 = nt0 << 4;

    const int m    = lane & 15;
    const int half = lane >> 4;
    const int KSTEPS = (Cin + 31) >> 5;
    const int NT     = (Cout + 15) >> 4;
    const bool act1  = (col0 + 16 < Cout);
    const bool rowOk = (row0 + m) < N_out;
    const int  chunkOff = half << 3;          // element offset of per-lane K chunk

    v8f acc0 = {0.f,0.f,0.f,0.f,0.f,0.f,0.f,0.f};
    v8f acc1 = {0.f,0.f,0.f,0.f,0.f,0.f,0.f,0.f};

    for (int k = 0; k < 27; ++k) {
        const int src = rowOk ? nbr[(size_t)k * N_out + row0 + m] : -1;
        const unsigned short* xrow = Xb + (size_t)src * Cin;

        for (int cc = 0; cc < KSTEPS; ++cc) {
            const int cbase = cc << 5;

            U32x8 u = { {0u,0u,0u,0u}, {0u,0u,0u,0u} };
            if (src >= 0) {
                if (Cin >= 16) {
                    u.lo = *(const uint4*)(xrow + cbase + chunkOff);          // K 0..15 half
                    if (cbase + 16 < Cin)
                        u.hi = *(const uint4*)(xrow + cbase + 16 + chunkOff); // K 16..31 half
                } else {
                    // Cin == 1: only K==0 (element 0 of half 0) is real data
                    if (half == 0) u.lo.x = (unsigned int)Xb[src];
                }
            }
            const v16us a_us = __builtin_bit_cast(v16us, u);

            const size_t bbase = ((size_t)k * KSTEPS + cc) * NT;
            v16us b0 = *(const v16us*)(Wb + (bbase + nt0) * 512 + lane * 16);
            acc0 = __builtin_amdgcn_wmma_f32_16x16x32_bf16(
                       false, __builtin_bit_cast(v16bf, a_us),
                       false, __builtin_bit_cast(v16bf, b0),
                       (short)0, acc0, false, false);
            if (act1) {
                v16us b1 = *(const v16us*)(Wb + (bbase + nt0 + 1) * 512 + lane * 16);
                acc1 = __builtin_amdgcn_wmma_f32_16x16x32_bf16(
                           false, __builtin_bit_cast(v16bf, a_us),
                           false, __builtin_bit_cast(v16bf, b1),
                           (short)0, acc1, false, false);
            }
        }
    }

    #pragma unroll
    for (int r = 0; r < 8; ++r) {            // D layout: VGPR r -> row r + 8*half
        int mm = row0 + r + (half << 3);
        if (mm < N_out) {
            Y[(size_t)mm * Cout + col0 + m] = acc0[r];
            if (act1) Y[(size_t)mm * Cout + col0 + 16 + m] = acc1[r];
        }
    }
}

// ---------------------------------------------------------------------------
// Utility / BN / pooling / heads
// ---------------------------------------------------------------------------
__global__ void fill_zero(float* p, int n) {
    int i = blockIdx.x * blockDim.x + threadIdx.x;
    if (i < n) p[i] = 0.f;
}

__global__ void bn_stats(const float* __restrict__ X, float* __restrict__ stats,
                         int N, int C, int rowsPerBlk) {
    int r0 = blockIdx.x * rowsPerBlk;
    int r1 = r0 + rowsPerBlk; if (r1 > N) r1 = N;
    for (int c = threadIdx.x; c < C; c += blockDim.x) {
        float s = 0.f, q = 0.f;
        for (int r = r0; r < r1; ++r) {
            float v = X[(size_t)r * C + c];
            s += v; q += v * v;
        }
        atomicAdd(&stats[c], s);
        atomicAdd(&stats[C + c], q);
    }
}

// normalize + ELU, emit bf16 activations for the next conv
__global__ void bn_elu_bf16(const float* __restrict__ X, const float* __restrict__ stats,
                            unsigned short* __restrict__ out, int N, int C) {
    size_t gid = (size_t)blockIdx.x * blockDim.x + threadIdx.x;
    size_t total = (size_t)N * C;
    if (gid >= total) return;
    int c = (int)(gid % C);
    float invN = 1.0f / (float)N;
    float mu  = stats[c] * invN;
    float var = stats[C + c] * invN - mu * mu;
    float y = (X[gid] - mu) * rsqrtf(var + 1e-5f);
    y = (y > 0.f) ? y : (expf(y) - 1.f);
    out[gid] = f32_to_bf16_rne(y);
}

__global__ void pool_accum(const unsigned short* __restrict__ H, const int* __restrict__ bidx,
                           float* __restrict__ pooled, float* __restrict__ cnt,
                           int N, int D) {
    size_t gid = (size_t)blockIdx.x * blockDim.x + threadIdx.x;
    if (gid >= (size_t)N * D) return;
    int n = (int)(gid / D);
    int d = (int)(gid % D);
    int b = bidx[n];
    atomicAdd(&pooled[(size_t)b * D + d], bf16_to_f32(H[gid]));
    if (d == 0) atomicAdd(&cnt[b], 1.0f);
}

// out = concat(mean(B,D), log_var(B,D)); gid = head*B*D + b*D + j
__global__ void heads(const float* __restrict__ pooled, const float* __restrict__ cnt,
                      const float* __restrict__ Wm, const float* __restrict__ bm,
                      const float* __restrict__ Wv, const float* __restrict__ bv,
                      float* __restrict__ out, int B, int D) {
    int gid = blockIdx.x * blockDim.x + threadIdx.x;
    if (gid >= 2 * B * D) return;
    int head = gid / (B * D);
    int b    = (gid / D) % B;
    int j    = gid % D;
    const float* W    = head ? Wv : Wm;
    const float* bias = head ? bv : bm;
    float inv = 1.0f / cnt[b];
    float s = 0.f;
    for (int i = 0; i < D; ++i)
        s += pooled[(size_t)b * D + i] * inv * W[(size_t)i * D + j];
    out[gid] = s + bias[j];
}

// ---------------------------------------------------------------------------
// Host driver.  Input order: [0] x ; [1..14] W1a..W7b ; [15] Wm [16] bm
// [17] Wv [18] bv ; [19..32] n1a..n7b ; [33] batch_idx
// ---------------------------------------------------------------------------
extern "C" void kernel_launch(void* const* d_in, const int* in_sizes, int n_in,
                              void* d_out, int out_size, void* d_ws, size_t ws_size,
                              hipStream_t stream) {
    (void)n_in; (void)ws_size;
    static const int CIN[14]  = {1,16, 16,32, 32,64, 64,128, 128,256, 256,512, 512,1024};
    static const int COUT[14] = {16,16, 32,32, 64,64, 128,128, 256,256, 512,512, 1024,1024};

    const float* Wc[14];
    const int*   NB[14];
    int          NOUT[14];
    for (int j = 0; j < 14; ++j) {
        Wc[j]   = (const float*)d_in[1 + j];
        NB[j]   = (const int*)d_in[19 + j];
        NOUT[j] = in_sizes[19 + j] / 27;
    }
    const float* Wm = (const float*)d_in[15];
    const float* bm = (const float*)d_in[16];
    const float* Wv = (const float*)d_in[17];
    const float* bv = (const float*)d_in[18];
    const int*   bidx = (const int*)d_in[33];

    const int D  = 1024;
    const int Bn = out_size / (2 * D);
    const int N0 = in_sizes[0];

    // Workspace sizing
    size_t maxConvB = 0, maxActE = (size_t)N0, maxWE = 0;
    for (int j = 0; j < 14; ++j) {
        size_t e = (size_t)NOUT[j] * COUT[j];
        if (e * 4 > maxConvB) maxConvB = e * 4;
        if (e > maxActE)      maxActE = e;
        size_t ks = (size_t)((CIN[j] + 31) / 32);
        size_t nt = (size_t)((COUT[j] + 15) / 16);
        size_t we = 27 * ks * nt * 512;
        if (we > maxWE) maxWE = we;
    }
    auto alignUp = [](size_t v) { return (v + 255) & ~(size_t)255; };
    size_t convBytes = alignUp(maxConvB);
    size_t actBytes  = alignUp(maxActE * 2);
    size_t wBytes    = alignUp(maxWE * 2);

    char* ws = (char*)d_ws;
    float*          convOut = (float*)ws;                            ws += convBytes;
    unsigned short* actBuf[2] = { (unsigned short*)ws,
                                  (unsigned short*)(ws + actBytes) }; ws += 2 * actBytes;
    unsigned short* wBuf = (unsigned short*)ws;                      ws += wBytes;
    float* stats  = (float*)ws;
    float* pooled = stats + 2048;
    float* cnt    = pooled + (size_t)Bn * D;

    // x -> bf16 into actBuf[0]
    f32_to_bf16_vec<<<(N0 + 255) / 256, 256, 0, stream>>>((const float*)d_in[0], actBuf[0], N0);

    int curIdx = 0;
    for (int j = 0; j < 14; ++j) {
        const int No = NOUT[j], Ci = CIN[j], Co = COUT[j];
        const int KS = (Ci + 31) / 32, NT = (Co + 15) / 16;
        const int outIdx = curIdx ^ 1;

        long long wtot = (long long)27 * KS * NT * 512;
        prep_weights<<<(unsigned)((wtot + 255) / 256), 256, 0, stream>>>(
            Wc[j], wBuf, Ci, Co, KS, NT, wtot);

        const int rowTiles  = (No + 15) / 16;
        const int colGroups = (Co + 31) / 32;
        const long long waves = (long long)rowTiles * colGroups;
        sparse_conv_wmma<<<(unsigned)((waves + 7) / 8), 256, 0, stream>>>(
            actBuf[curIdx], wBuf, NB[j], convOut, No, Ci, Co, rowTiles, colGroups);

        fill_zero<<<(2 * Co + 255) / 256, 256, 0, stream>>>(stats, 2 * Co);
        const int rowsPerBlk = 512;
        bn_stats<<<(No + rowsPerBlk - 1) / rowsPerBlk, 256, 0, stream>>>(convOut, stats, No, Co, rowsPerBlk);
        size_t tot = (size_t)No * Co;
        bn_elu_bf16<<<(unsigned)((tot + 255) / 256), 256, 0, stream>>>(
            convOut, stats, actBuf[outIdx], No, Co);

        curIdx = outIdx;
    }

    const int N7 = NOUT[13];
    fill_zero<<<(Bn * D + Bn + 255) / 256, 256, 0, stream>>>(pooled, Bn * D + Bn);
    size_t ptot = (size_t)N7 * D;
    pool_accum<<<(unsigned)((ptot + 255) / 256), 256, 0, stream>>>(actBuf[curIdx], bidx, pooled, cnt, N7, D);
    heads<<<(2 * Bn * D + 255) / 256, 256, 0, stream>>>(pooled, cnt, Wm, bm, Wv, bv, (float*)d_out, Bn, D);
}
